// HeteroGNN_43078521978921
// MI455X (gfx1250) — matrix operand order repaired and checked by reading, not compiled, not run on hardware
//
#include <hip/hip_runtime.h>
#include <hip/hip_bf16.h>
#include <cstdint>
#include <cstddef>

// Problem constants (match reference)
#define NN   50000
#define EE   500000
#define INC  128
#define HH   2
#define CC   128
#define HC   256            // HH*CC
#define ENN  (EE + NN)      // edges + self loops
#define SLOPE 0.2f

typedef __attribute__((ext_vector_type(16))) _Float16 v16h;
typedef __attribute__((ext_vector_type(8)))  _Float16 v8h;
typedef __attribute__((ext_vector_type(8)))  float    v8f;

__device__ __forceinline__ float leaky(float v) { return v > 0.f ? v : SLOPE * v; }

// Monotone-bit float atomic max: int-compare for >=0, uint-min for <0.
__device__ __forceinline__ void atomicMaxF(float* addr, float val) {
  if (val >= 0.f) atomicMax((int*)addr, __float_as_int(val));
  else            atomicMin((unsigned int*)addr, __float_as_uint(val));
}

// ---------------- conversion / init ----------------
__global__ void k_cvt_x(const float* __restrict__ x, _Float16* __restrict__ xh) {
  int i = blockIdx.x * blockDim.x + threadIdx.x;
  if (i < NN * INC) xh[i] = (_Float16)x[i];
}

// Pack W[128x256] (row-major, K x N) into WMMA-B fragment order:
// bp[((nt*4+kt)*32 + lane)*16 + e] = W[kt*32 + (lane>>4)*16 + e][nt*16 + (lane&15)]
__global__ void k_pack_w(const float* __restrict__ W, _Float16* __restrict__ bp) {
  int t = blockIdx.x * blockDim.x + threadIdx.x;
  if (t >= INC * HC) return;
  int e    = t & 15;
  int lane = (t >> 4) & 31;
  int kt   = (t >> 9) & 3;
  int nt   = t >> 11;
  int K    = kt * 32 + (lane >> 4) * 16 + e;
  int col  = nt * 16 + (lane & 15);
  bp[t] = (_Float16)W[K * HC + col];
}

__global__ void k_init_out(float* __restrict__ out, const float* __restrict__ b0,
                           const float* __restrict__ b1, const float* __restrict__ b2) {
  int i = blockIdx.x * blockDim.x + threadIdx.x;
  if (i < NN * HC) { int j = i & (HC - 1); out[i] = b0[j] + b1[j] + b2[j]; }
}

__global__ void k_init_md(float* __restrict__ m, float* __restrict__ d) {
  int i = blockIdx.x * blockDim.x + threadIdx.x;
  if (i < NN * HH) { m[i] = -INFINITY; d[i] = 0.f; }
}

// ---------------- WMMA GEMM: xl[N,256] = xh[N,128] @ Wpacked ----------------
// Block = 16 waves; block owns rows [m0, m0+16); wave w owns cols [16w, 16w+16).
__global__ __launch_bounds__(512) void k_gemm(const _Float16* __restrict__ xh,
                                              const _Float16* __restrict__ bp,
                                              float* __restrict__ xl) {
  int m0   = blockIdx.x * 16;
  int wave = threadIdx.x >> 5;     // 0..15 == n-tile
  int lane = threadIdx.x & 31;
  int mr   = lane & 15;            // A row within tile
  int kh   = lane >> 4;            // lane half selects K sub-range
  const _Float16* arow = xh + (size_t)(m0 + mr) * INC + kh * 16;
  const v16h* bfr = (const v16h*)bp + wave * 128 + lane;   // + kt*32 per step
  v8f acc = {};
#if __has_builtin(__builtin_amdgcn_wmma_f32_16x16x32_f16)
  #pragma unroll
  for (int kt = 0; kt < 4; ++kt) {
    // A k order used here (contiguous, same as packed B): elems 0..7 -> K = kb + kh*16 + 0..7,
    //                        elems 8..15 -> K = kb + kh*16 + 8..15
    v8h lo = *(const v8h*)(arow + kt * 32);
    v8h hi = *(const v8h*)(arow + kt * 32 + 8);
    v16h a = __builtin_shufflevector(lo, hi, 0,1,2,3,4,5,6,7,8,9,10,11,12,13,14,15);
    v16h b = bfr[kt * 32];
    acc = __builtin_amdgcn_wmma_f32_16x16x32_f16(false, a, false, b,
                                                 (short)0, acc, false, false);
  }
#else
  // Scalar fallback (same packed layout), keeps file compiling on any toolchain.
  #pragma unroll
  for (int v = 0; v < 8; ++v) {
    int M = v + 8 * kh;
    float s = 0.f;
    for (int k = 0; k < INC; ++k) {
      int kt = k >> 5, rem = k & 31, lh = rem >> 4, e = rem & 15;
      float bv = (float)bp[wave * 2048 + kt * 512 + (lh * 16 + mr) * 16 + e];
      s += (float)xh[(size_t)(m0 + M) * INC + k] * bv;
    }
    acc[v] = s;
  }
#endif
  // D layout: VGPR v, lanes 0-15 -> M=v ; lanes 16-31 -> M=v+8 ; N = lane&15
  #pragma unroll
  for (int v = 0; v < 8; ++v) {
    int M = v + 8 * kh;
    xl[(size_t)(m0 + M) * HC + wave * 16 + mr] = acc[v];
  }
}

// ---------------- attention logits: one wave per (node, head) ----------------
__global__ void k_att(const float* __restrict__ xl, const float* __restrict__ asv,
                      const float* __restrict__ adv, float* __restrict__ a_src,
                      float* __restrict__ a_dst) {
  int w    = (blockIdx.x * blockDim.x + threadIdx.x) >> 5;
  int lane = threadIdx.x & 31;
  if (w >= NN * HH) return;
  int n = w >> 1, h = w & 1;
  const float* row = xl + (size_t)n * HC + h * CC;
  const float* s0  = asv + h * CC;
  const float* d0  = adv + h * CC;
  float s = 0.f, d = 0.f;
  #pragma unroll
  for (int i = 0; i < 4; ++i) {
    float v = row[lane + i * 32];
    s += v * s0[lane + i * 32];
    d += v * d0[lane + i * 32];
  }
  #pragma unroll
  for (int off = 16; off > 0; off >>= 1) {   // wave32 reduction
    s += __shfl_xor(s, off, 32);
    d += __shfl_xor(d, off, 32);
  }
  if (lane == 0) { a_src[w] = s; a_dst[w] = d; }
}

// ---------------- edge passes ----------------
__device__ __forceinline__ void edge_ids(int t, const int* __restrict__ ei,
                                         int& src, int& dst) {
  if (t < EE) { src = ei[t]; dst = ei[EE + t]; }
  else        { src = dst = t - EE; }          // self loop
}

__global__ void k_edge_max(const int* __restrict__ ei, const float* __restrict__ a_src,
                           const float* __restrict__ a_dst, float* __restrict__ m) {
  int t = blockIdx.x * blockDim.x + threadIdx.x;
  if (t >= ENN) return;
  int src, dst; edge_ids(t, ei, src, dst);
  #pragma unroll
  for (int h = 0; h < HH; ++h)
    atomicMaxF(&m[dst * HH + h], leaky(a_src[src * HH + h] + a_dst[dst * HH + h]));
}

__global__ void k_edge_sum(const int* __restrict__ ei, const float* __restrict__ a_src,
                           const float* __restrict__ a_dst, const float* __restrict__ m,
                           float* __restrict__ den) {
  int t = blockIdx.x * blockDim.x + threadIdx.x;
  if (t >= ENN) return;
  int src, dst; edge_ids(t, ei, src, dst);
  #pragma unroll
  for (int h = 0; h < HH; ++h) {
    float v = leaky(a_src[src * HH + h] + a_dst[dst * HH + h]);
    atomicAdd(&den[dst * HH + h], expf(v - m[dst * HH + h]));
  }
}

// One wave per (edge, head); lanes cover 128 channels in 4 coalesced strides.
__global__ void k_edge_scatter(const int* __restrict__ ei, const float* __restrict__ a_src,
                               const float* __restrict__ a_dst, const float* __restrict__ m,
                               const float* __restrict__ den, const float* __restrict__ xl,
                               float* __restrict__ out) {
  long long g = (long long)blockIdx.x * blockDim.x + threadIdx.x;
  int w    = (int)(g >> 5);
  int lane = (int)(g & 31);
  if (w >= ENN * HH) return;
  int t = w >> 1, h = w & 1;
  int src, dst; edge_ids(t, ei, src, dst);
  float logit = leaky(a_src[src * HH + h] + a_dst[dst * HH + h]);
  float alpha = expf(logit - m[dst * HH + h]) / den[dst * HH + h];
  const float* xr = xl + (size_t)src * HC + h * CC;
  float* orow     = out + (size_t)dst * HC + h * CC;
  #pragma unroll
  for (int i = 0; i < 4; ++i) {
    int c = lane + i * 32;
    atomicAdd(&orow[c], xr[c] * alpha);
  }
}

// ---------------- host side ----------------
extern "C" void kernel_launch(void* const* d_in, const int* in_sizes, int n_in,
                              void* d_out, int out_size, void* d_ws, size_t ws_size,
                              hipStream_t stream) {
  (void)in_sizes; (void)n_in; (void)out_size; (void)ws_size;

  const float* x = (const float*)d_in[0];
  const int* ei[3] = {(const int*)d_in[1], (const int*)d_in[2], (const int*)d_in[3]};
  const float *W[3], *AS[3], *AD[3], *B[3];
  for (int r = 0; r < 3; ++r) {
    W[r]  = (const float*)d_in[4 + 4 * r];
    AS[r] = (const float*)d_in[5 + 4 * r];
    AD[r] = (const float*)d_in[6 + 4 * r];
    B[r]  = (const float*)d_in[7 + 4 * r];
  }
  float* out = (float*)d_out;

  // Workspace carve-up (~66 MB total)
  char* p = (char*)d_ws;
  _Float16* xh = (_Float16*)p; p += (size_t)NN * INC * sizeof(_Float16);  // 12.8 MB
  _Float16* bp = (_Float16*)p; p += (size_t)INC * HC * sizeof(_Float16);  // 64 KB
  float* xl    = (float*)p;    p += (size_t)NN * HC * sizeof(float);      // 51.2 MB
  float* a_src = (float*)p;    p += (size_t)NN * HH * sizeof(float);
  float* a_dst = (float*)p;    p += (size_t)NN * HH * sizeof(float);
  float* m     = (float*)p;    p += (size_t)NN * HH * sizeof(float);
  float* den   = (float*)p;    p += (size_t)NN * HH * sizeof(float);

  const int TB = 256;
  k_cvt_x<<<(NN * INC + TB - 1) / TB, TB, 0, stream>>>(x, xh);
  k_init_out<<<(NN * HC + TB - 1) / TB, TB, 0, stream>>>(out, B[0], B[1], B[2]);

  for (int r = 0; r < 3; ++r) {
    k_pack_w<<<(INC * HC + TB - 1) / TB, TB, 0, stream>>>(W[r], bp);
    k_gemm<<<NN / 16, 512, 0, stream>>>(xh, bp, xl);
    k_att<<<(NN * HH * 32 + TB - 1) / TB, TB, 0, stream>>>(xl, AS[r], AD[r], a_src, a_dst);
    k_init_md<<<(NN * HH + TB - 1) / TB, TB, 0, stream>>>(m, den);
    k_edge_max<<<(ENN + TB - 1) / TB, TB, 0, stream>>>(ei[r], a_src, a_dst, m);
    k_edge_sum<<<(ENN + TB - 1) / TB, TB, 0, stream>>>(ei[r], a_src, a_dst, m, den);
    long long thr = (long long)ENN * HH * 32;
    k_edge_scatter<<<(unsigned)((thr + TB - 1) / TB), TB, 0, stream>>>(
        ei[r], a_src, a_dst, m, den, xl, out);
  }
}
